// ProxyNCA_distribution_loss_19061064860142
// MI455X (gfx1250) — compile-verified, matrix operand order checked by reading, not run on hardware
//
#include <hip/hip_runtime.h>

typedef __attribute__((ext_vector_type(16))) __bf16 v16bf;
typedef __attribute__((ext_vector_type(8)))  __bf16 v8bf;
typedef __attribute__((ext_vector_type(8)))  float  v8f;
typedef __attribute__((ext_vector_type(4)))  int    v4i;

#define NROW 4096
#define NCLS 8192
#define DEMB 512
#define KTOT 1024          // [x^2 | x] concat
#define KSTEP 32
#define NCHUNK 8
#define C_CHUNK (NCLS / NCHUNK)   // 1024
#define LDS_PITCH 40              // 32 bf16 + 8 pad (16B) -> conflict-free b128 reads

#if defined(__has_builtin)
#if __has_builtin(__builtin_amdgcn_global_load_async_to_lds_b128)
#define HAVE_ASYNC_LDS 1
#endif
#endif
#ifndef HAVE_ASYNC_LDS
#define HAVE_ASYNC_LDS 0
#endif

#if HAVE_ASYNC_LDS
typedef __attribute__((address_space(3))) v4i lds_v4i_t;   // LDS dest: v4i addrspace(3)*
typedef __attribute__((address_space(1))) v4i gbl_v4i_t;   // global src: v4i addrspace(1)*
#endif

__device__ __forceinline__ void wait_async_all() {
#if HAVE_ASYNC_LDS
#if defined(__has_builtin) && __has_builtin(__builtin_amdgcn_s_wait_asynccnt)
    __builtin_amdgcn_s_wait_asynccnt(0);
#else
    asm volatile("s_wait_asynccnt 0" ::: "memory");
#endif
#endif
}

__device__ __forceinline__ float softplus_f(float v) {
    return (v > 20.0f) ? v : log1pf(__expf(v));
}

__global__ void init_kernel(float* klAcc) {
    if (threadIdx.x == 0 && blockIdx.x == 0) klAcc[0] = 0.0f;
}

// One workgroup per class c: normalize proxy row, build packed B^T row (bf16),
// p2s[c], and KL contribution.
__global__ __launch_bounds__(256) void prep_proxies_kernel(
    const float* __restrict__ proxies, const float* __restrict__ sigmas_inv,
    __bf16* __restrict__ Bt, float* __restrict__ p2s, float* __restrict__ klAcc) {
    __shared__ float red[256];
    const int c = blockIdx.x, tid = threadIdx.x;
    const float* pr = proxies    + (size_t)c * DEMB;
    const float* sg = sigmas_inv + (size_t)c * DEMB;
    float p0 = pr[tid], p1 = pr[tid + 256];
    red[tid] = p0 * p0 + p1 * p1;
    __syncthreads();
    for (int s = 128; s > 0; s >>= 1) { if (tid < s) red[tid] += red[tid + s]; __syncthreads(); }
    const float inv = 1.0f / fmaxf(sqrtf(red[0]), 1e-12f);
    __syncthreads();

    float p2sv = 0.0f, klv = 0.0f;
    #pragma unroll
    for (int h = 0; h < 2; ++h) {
        const int d = tid + h * 256;
        const float p  = (h ? p1 : p0);
        const float pn = p * inv;          // normalized proxy
        const float Pv = 3.0f * pn;        // SCALE * normalized
        const float sp = softplus_f(sg[d]);
        const float s2 = sp * sp;          // s[c,d]
        Bt[(size_t)c * KTOT + d]        = (__bf16)s2;
        Bt[(size_t)c * KTOT + DEMB + d] = (__bf16)(-2.0f * Pv * s2);
        p2sv += Pv * Pv * s2;
        klv  += 0.5f * (1.0f / s2 + pn * pn - 1.0f + __logf(s2));
    }
    red[tid] = p2sv; __syncthreads();
    for (int s = 128; s > 0; s >>= 1) { if (tid < s) red[tid] += red[tid + s]; __syncthreads(); }
    const float p2tot = red[0];
    __syncthreads();
    red[tid] = klv; __syncthreads();
    for (int s = 128; s > 0; s >>= 1) { if (tid < s) red[tid] += red[tid + s]; __syncthreads(); }
    if (tid == 0) { p2s[c] = p2tot; atomicAdd(klAcc, red[0]); }
}

// One workgroup per sample n: build packed A row [xn^2 | xn] in bf16.
__global__ __launch_bounds__(256) void prep_x_kernel(
    const float* __restrict__ X, __bf16* __restrict__ A) {
    __shared__ float red[256];
    const int n = blockIdx.x, tid = threadIdx.x;
    const float* xr = X + (size_t)n * DEMB;
    float x0 = xr[tid], x1 = xr[tid + 256];
    red[tid] = x0 * x0 + x1 * x1;
    __syncthreads();
    for (int s = 128; s > 0; s >>= 1) { if (tid < s) red[tid] += red[tid + s]; __syncthreads(); }
    const float inv = 3.0f / fmaxf(sqrtf(red[0]), 1e-12f);
    #pragma unroll
    for (int h = 0; h < 2; ++h) {
        const int d = tid + h * 256;
        const float xn = (h ? x1 : x0) * inv;
        A[(size_t)n * KTOT + d]        = (__bf16)(xn * xn);
        A[(size_t)n * KTOT + DEMB + d] = (__bf16)xn;
    }
}

// Issue one k-step's A (64x32) and B (128x32) tile fill into LDS.
// Async path: GLOBAL_LOAD_ASYNC_TO_LDS_B128 (ASYNCcnt); fallback: VGPR staging.
__device__ __forceinline__ void stage_tiles(
    const __bf16* __restrict__ A, const __bf16* __restrict__ Bt,
    __bf16* as, __bf16* bs, int rowBase, int cTile, int k, int tid) {
    const int rA = tid >> 2, ccA = tid & 3;
#if HAVE_ASYNC_LDS
    __builtin_amdgcn_global_load_async_to_lds_b128(
        (gbl_v4i_t*)(A + (size_t)(rowBase + rA) * KTOT + k + ccA * 8),
        (lds_v4i_t*)(&as[rA * LDS_PITCH + ccA * 8]), 0, 0);
    #pragma unroll
    for (int it = 0; it < 2; ++it) {
        const int idx = tid + it * 256;
        const int r = idx >> 2, cc = idx & 3;
        __builtin_amdgcn_global_load_async_to_lds_b128(
            (gbl_v4i_t*)(Bt + (size_t)(cTile + r) * KTOT + k + cc * 8),
            (lds_v4i_t*)(&bs[r * LDS_PITCH + cc * 8]), 0, 0);
    }
#else
    {
        const int4* src = (const int4*)(A + (size_t)(rowBase + rA) * KTOT + k + ccA * 8);
        *(int4*)(&as[rA * LDS_PITCH + ccA * 8]) = *src;
    }
    #pragma unroll
    for (int it = 0; it < 2; ++it) {
        const int idx = tid + it * 256;
        const int r = idx >> 2, cc = idx & 3;
        const int4* src = (const int4*)(Bt + (size_t)(cTile + r) * KTOT + k + cc * 8);
        *(int4*)(&bs[r * LDS_PITCH + cc * 8]) = *src;
    }
#endif
}

// Fused WMMA GEMM + sum(exp(-D)) epilogue, double-buffered LDS.
// Grid: (NROW/64) * NCHUNK workgroups, 256 threads (8 waves).
__global__ __launch_bounds__(256) void gemm_lse_kernel(
    const __bf16* __restrict__ A, const __bf16* __restrict__ Bt,
    const float* __restrict__ p2s, float* __restrict__ psum) {
    __shared__ __bf16 As[2][64 * LDS_PITCH];
    __shared__ __bf16 Bs[2][128 * LDS_PITCH];
    __shared__ float rowsum[64];

    const int tid  = threadIdx.x;
    const int wid  = tid >> 5, lane = tid & 31;
    const int rowBlock = blockIdx.x / NCHUNK;
    const int chunk    = blockIdx.x % NCHUNK;
    const int rowBase  = rowBlock * 64;
    const int strip    = wid >> 1;        // 0..3 : 16-row strip
    const int colHalf  = wid & 1;         // 0..1 : 64-col half
    const int l16      = lane & 15;
    const int kSel     = (lane & 16) ? 8 : 0;   // ISA bf16 A/B frag K-selection

    if (tid < 64) rowsum[tid] = 0.0f;

    float ssum[8];
    #pragma unroll
    for (int j = 0; j < 8; ++j) ssum[j] = 0.0f;

    for (int cTile = chunk * C_CHUNK; cTile < (chunk + 1) * C_CHUNK; cTile += 128) {
        v8f acc[4] = {};
        // prologue: fill buffer 0 with k=0
        stage_tiles(A, Bt, As[0], Bs[0], rowBase, cTile, 0, tid);
        wait_async_all();
        __syncthreads();
        int cur = 0;
        for (int k = 0; k < KTOT; k += KSTEP) {
            // prefetch next k-step into the other buffer (safe: last read of it
            // finished at the barrier ending iteration k-1)
            if (k + KSTEP < KTOT)
                stage_tiles(A, Bt, As[cur ^ 1], Bs[cur ^ 1], rowBase, cTile, k + KSTEP, tid);

            // A fragment (16x32 strip), per ISA bf16 layout
            v16bf a;
            {
                const __bf16* asb = As[cur];
                const v8bf lo = *(const v8bf*)(&asb[(strip * 16 + l16) * LDS_PITCH + kSel]);
                const v8bf hi = *(const v8bf*)(&asb[(strip * 16 + l16) * LDS_PITCH + kSel + 16]);
                #pragma unroll
                for (int i = 0; i < 8; ++i) { a[i] = lo[i]; a[i + 8] = hi[i]; }
            }
            #pragma unroll
            for (int t = 0; t < 4; ++t) {
                const int cb = colHalf * 64 + t * 16;
                const __bf16* bsb = Bs[cur];
                const v8bf lo = *(const v8bf*)(&bsb[(cb + l16) * LDS_PITCH + kSel]);
                const v8bf hi = *(const v8bf*)(&bsb[(cb + l16) * LDS_PITCH + kSel + 16]);
                v16bf b;
                #pragma unroll
                for (int i = 0; i < 8; ++i) { b[i] = lo[i]; b[i + 8] = hi[i]; }
                acc[t] = __builtin_amdgcn_wmma_f32_16x16x32_bf16(
                    false, a, false, b, (short)0, acc[t], false, false);
            }
            wait_async_all();
            __syncthreads();
            cur ^= 1;
        }
        // epilogue: D = acc + p2s[c]; accumulate exp(-D). D>=0 so no max needed.
        #pragma unroll
        for (int t = 0; t < 4; ++t) {
            const float pv = p2s[cTile + colHalf * 64 + t * 16 + l16];
            #pragma unroll
            for (int j = 0; j < 8; ++j)
                ssum[j] += __expf(-(acc[t][j] + pv));
        }
    }
    // combine lanes/waves per row via LDS float atomics (ds_add_f32)
    #pragma unroll
    for (int j = 0; j < 8; ++j) {
        const int r = strip * 16 + j + ((lane & 16) ? 8 : 0);
        atomicAdd(&rowsum[r], ssum[j]);
    }
    __syncthreads();
    if (tid < 64) psum[(size_t)chunk * NROW + rowBase + tid] = rowsum[tid];
}

// D[n, T[n]] from the same packed bf16 operands (consistent with GEMM).
__global__ __launch_bounds__(256) void dtgt_kernel(
    const __bf16* __restrict__ A, const __bf16* __restrict__ Bt,
    const float* __restrict__ p2s, const int* __restrict__ T,
    float* __restrict__ Dtgt) {
    __shared__ float red[256];
    const int n = blockIdx.x, tid = threadIdx.x;
    const int t = T[n];
    const __bf16* ar = A  + (size_t)n * KTOT;
    const __bf16* br = Bt + (size_t)t * KTOT;
    float dot = 0.0f;
    for (int k = tid; k < KTOT; k += 256) dot += (float)ar[k] * (float)br[k];
    red[tid] = dot; __syncthreads();
    for (int s = 128; s > 0; s >>= 1) { if (tid < s) red[tid] += red[tid + s]; __syncthreads(); }
    if (tid == 0) Dtgt[n] = red[0] + p2s[t];
}

__global__ __launch_bounds__(256) void finalize_kernel(
    const float* __restrict__ psum, const float* __restrict__ Dtgt,
    const float* __restrict__ klAcc, float* __restrict__ out) {
    __shared__ float red[256];
    const int tid = threadIdx.x;
    float ce = 0.0f;
    for (int n = tid; n < NROW; n += 256) {
        float s = 0.0f;
        #pragma unroll
        for (int ch = 0; ch < NCHUNK; ++ch) s += psum[(size_t)ch * NROW + n];
        ce += Dtgt[n] + __logf(s);   // -logp[n,T]
    }
    red[tid] = ce; __syncthreads();
    for (int s = 128; s > 0; s >>= 1) { if (tid < s) red[tid] += red[tid + s]; __syncthreads(); }
    if (tid == 0)
        out[0] = red[0] / (float)NROW + 0.2f * (klAcc[0] / (float)NCLS);
}

extern "C" void kernel_launch(void* const* d_in, const int* in_sizes, int n_in,
                              void* d_out, int out_size, void* d_ws, size_t ws_size,
                              hipStream_t stream) {
    const float* X       = (const float*)d_in[0];
    // d_in[1] = indices (unused by the reference)
    const int*   T       = (const int*)d_in[2];
    const float* proxies = (const float*)d_in[3];
    const float* sigmas  = (const float*)d_in[4];
    float* out = (float*)d_out;

    char* w = (char*)d_ws;
    __bf16* A    = (__bf16*)w; w += (size_t)NROW * KTOT * sizeof(__bf16);   // 8 MB
    __bf16* Bt   = (__bf16*)w; w += (size_t)NCLS * KTOT * sizeof(__bf16);   // 16 MB
    float*  p2s  = (float*)w;  w += (size_t)NCLS * sizeof(float);
    float*  psum = (float*)w;  w += (size_t)NCHUNK * NROW * sizeof(float);
    float*  Dtgt = (float*)w;  w += (size_t)NROW * sizeof(float);
    float*  klAcc = (float*)w; w += 256;

    init_kernel<<<1, 1, 0, stream>>>(klAcc);
    prep_proxies_kernel<<<NCLS, 256, 0, stream>>>(proxies, sigmas, Bt, p2s, klAcc);
    prep_x_kernel<<<NROW, 256, 0, stream>>>(X, A);
    gemm_lse_kernel<<<(NROW / 64) * NCHUNK, 256, 0, stream>>>(A, Bt, p2s, psum);
    dtgt_kernel<<<NROW, 256, 0, stream>>>(A, Bt, p2s, T, Dtgt);
    finalize_kernel<<<1, 256, 0, stream>>>(psum, Dtgt, klAcc, out);
}
